// AMNet_70300024701104
// MI455X (gfx1250) — compile-verified, matrix operand order checked by reading, not run on hardware
//
#include <hip/hip_runtime.h>
#include <hip/hip_bf16.h>

// ---------------------------------------------------------------------------
// AMNet-style GRU tracker for MI455X (gfx1250, wave32, WMMA bf16 16x16x32).
//
// Pipeline:
//   prep kernels : one-time f32 -> bf16 weight conversion (K-contiguous layout)
//   k_phi_gemm   : xv  = relu(flow[15872,4096] @ W_phi[4096,512] + b)   (WMMA)
//   k_gi0_gemm   : gi0 = x_t @ Wih0^T + bih0 for ALL t (no recurrence)  (WMMA)
//   k_recurrent  : 512 sequential steps on one persistent workgroup
//                  (32 waves, h-state bf16 in LDS, WMMA gate GEMMs)
// Workspace: xv f32 (32.5 MB) + gi0 f32 (94.4 MB) + bf16 weights (~12 MB).
// ---------------------------------------------------------------------------

#define T_STEPS 512
#define NBOX 30
#define XDIM 4096
#define HD 512
#define HC 32

typedef __bf16 bf16_t;
typedef bf16_t v16bf __attribute__((ext_vector_type(16)));
typedef bf16_t v8bf  __attribute__((ext_vector_type(8)));
typedef float  v8f   __attribute__((ext_vector_type(8)));

static __device__ __forceinline__ bf16_t f2bf(float f) {
    union { float f; unsigned int u; } a; a.f = f;
    unsigned int r = a.u + 0x7FFFu + ((a.u >> 16) & 1u);   // round-nearest-even
    union { unsigned short u; bf16_t b; } c; c.u = (unsigned short)(r >> 16);
    return c.b;
}
static __device__ __forceinline__ float bf2f(bf16_t x) {
    union { bf16_t b; unsigned short u; } c; c.b = x;
    union { unsigned int u; float f; } r; r.u = ((unsigned int)c.u) << 16;
    return r.f;
}
static __device__ __forceinline__ float sigm(float x) { return 1.0f / (1.0f + expf(-x)); }

static __device__ __forceinline__ v8f v8zero() {
    v8f z;
#pragma unroll
    for (int i = 0; i < 8; ++i) z[i] = 0.0f;
    return z;
}

static __device__ __forceinline__ v16bf cat16(v8bf lo, v8bf hi) {
    return __builtin_shufflevector(lo, hi, 0, 1, 2, 3, 4, 5, 6, 7,
                                   8, 9, 10, 11, 12, 13, 14, 15);
}

static __device__ __forceinline__ v8f wmma_bf16(v16bf a, v16bf b, v8f c) {
    return __builtin_amdgcn_wmma_f32_16x16x32_bf16(false, a, false, b, (short)0, c,
                                                   false, false);
}

// A fragment (16x32 bf16) from row-major f32 global:  lanes 0-15 hold K=0..7,16..23
// of row (lane&15); lanes 16-31 hold K=8..15,24..31.
static __device__ __forceinline__ v16bf load_a_g(const float* __restrict__ A, int ld,
                                                 int m0, int k0, int lane) {
    const float* p = A + (size_t)(m0 + (lane & 15)) * ld + (k0 + ((lane >> 4) << 3));
    v16bf a;
#pragma unroll
    for (int e = 0; e < 8; ++e) { a[e] = f2bf(p[e]); a[e + 8] = f2bf(p[e + 16]); }
    return a;
}

// B fragment where B(k, n) = W[n][k], W pre-converted bf16, K-contiguous rows.
// Two 16-byte global_load_b128 per fragment, zero conversion VALU.
static __device__ __forceinline__ v16bf load_b_bf(const bf16_t* __restrict__ W, int ldk,
                                                  int gc0, int k0, int lane) {
    const bf16_t* p = W + (size_t)(gc0 + (lane & 15)) * ldk + (k0 + ((lane >> 4) << 3));
    v8bf lo = *(const v8bf*)p;
    v8bf hi = *(const v8bf*)(p + 16);
    return cat16(lo, hi);
}

// A fragment from bf16 hidden state kept in LDS (row stride HD, 16B aligned).
static __device__ __forceinline__ v16bf load_a_lds(const bf16_t* h, int m0, int k0,
                                                   int lane) {
    const bf16_t* p = h + (m0 + (lane & 15)) * HD + (k0 + ((lane >> 4) << 3));
    v8bf lo = *(const v8bf*)p;
    v8bf hi = *(const v8bf*)(p + 16);
    return cat16(lo, hi);
}

// ---------------------------------------------------------------------------
// Prep kernels: one-time weight conversion to bf16.
// ---------------------------------------------------------------------------
// W_phi [K=4096][N=512] f32 -> WphiT [N=512][K=4096] bf16 (K-contiguous).
__global__ __launch_bounds__(256) void k_prep_wphiT(const float* __restrict__ W,
                                                    bf16_t* __restrict__ Wt) {
    const int idx = blockIdx.x * 256 + threadIdx.x;     // over N*K
    const int n = idx >> 12;                             // /4096
    const int k = idx & (XDIM - 1);
    Wt[(size_t)n * XDIM + k] = f2bf(W[(size_t)k * HD + n]);
}
// Generic row-major f32 -> bf16 copy (layout preserved).
__global__ __launch_bounds__(256) void k_prep_bf16(const float* __restrict__ src,
                                                   bf16_t* __restrict__ dst, int n) {
    const int idx = blockIdx.x * 256 + threadIdx.x;
    if (idx < n) dst[idx] = f2bf(src[idx]);
}

// ---------------------------------------------------------------------------
// Kernel 1: xv = relu(flow @ W_phi + b_phi).  M=15872, N=512, K=4096.
// Block = 8 waves, each wave owns a 16x64 tile (4 accumulators).
// ---------------------------------------------------------------------------
__global__ __launch_bounds__(256) void k_phi_gemm(const float* __restrict__ flow,
                                                  const bf16_t* __restrict__ WphiT,
                                                  const float* __restrict__ bphi,
                                                  float* __restrict__ xv) {
    const int lane = threadIdx.x & 31;
    const int w    = threadIdx.x >> 5;
    const int m0   = blockIdx.x * 128 + w * 16;
    const int n0   = blockIdx.y * 64;

    v8f acc[4];
#pragma unroll
    for (int j = 0; j < 4; ++j) acc[j] = v8zero();

    for (int k0 = 0; k0 < XDIM; k0 += 32) {
        v16bf a = load_a_g(flow, XDIM, m0, k0, lane);
#pragma unroll
        for (int j = 0; j < 4; ++j) {
            v16bf b = load_b_bf(WphiT, XDIM, n0 + 16 * j, k0, lane);
            acc[j] = wmma_bf16(a, b, acc[j]);
        }
    }
    const int mh = (lane >> 4) << 3;
#pragma unroll
    for (int j = 0; j < 4; ++j) {
        const int n = n0 + 16 * j + (lane & 15);
        const float bias = bphi[n];
#pragma unroll
        for (int i = 0; i < 8; ++i) {
            const int m = m0 + mh + i;
            float v = acc[j][i] + bias;
            xv[(size_t)m * HD + n] = v > 0.0f ? v : 0.0f;
        }
    }
}

// ---------------------------------------------------------------------------
// Kernel 2: gi0 = x_t @ Wih0^T + bih0 for all t.  M=15360, N=1536, K=1024.
// A row m = t*30+b gathers [xv[t,1+b,:], xv[t,0,:]]; 32-wide K blocks never
// straddle the 512 boundary, so the source row is a uniform per-iter select.
// ---------------------------------------------------------------------------
__global__ __launch_bounds__(256) void k_gi0_gemm(const float* __restrict__ xv,
                                                  const bf16_t* __restrict__ Wih0b,
                                                  const float* __restrict__ bih0,
                                                  float* __restrict__ gi0) {
    const int lane = threadIdx.x & 31;
    const int w    = threadIdx.x >> 5;
    const int m0   = blockIdx.x * 128 + w * 16;
    const int n0   = blockIdx.y * 64;

    const int r = m0 + (lane & 15);
    const int t = r / NBOX;
    const int b = r % NBOX;
    const float* rowA = xv + (size_t)(t * 31 + 1 + b) * HD; // track feature (k<512)
    const float* rowI = xv + (size_t)(t * 31) * HD;         // image feature (k>=512)

    v8f acc[4];
#pragma unroll
    for (int j = 0; j < 4; ++j) acc[j] = v8zero();

    for (int k0 = 0; k0 < 2 * HD; k0 += 32) {
        const float* rp = (k0 < HD) ? (rowA + k0) : (rowI + (k0 - HD));
        const float* p  = rp + ((lane >> 4) << 3);
        v16bf a;
#pragma unroll
        for (int e = 0; e < 8; ++e) { a[e] = f2bf(p[e]); a[e + 8] = f2bf(p[e + 16]); }
#pragma unroll
        for (int j = 0; j < 4; ++j) {
            v16bf bf = load_b_bf(Wih0b, 2 * HD, n0 + 16 * j, k0, lane);
            acc[j] = wmma_bf16(a, bf, acc[j]);
        }
    }
    const int mh = (lane >> 4) << 3;
#pragma unroll
    for (int j = 0; j < 4; ++j) {
        const int n = n0 + 16 * j + (lane & 15);
        const float bias = bih0[n];
#pragma unroll
        for (int i = 0; i < 8; ++i) {
            const int m = m0 + mh + i;
            gi0[(size_t)m * (3 * HD) + n] = acc[j][i] + bias;
        }
    }
}

// ---------------------------------------------------------------------------
// Kernel 3: persistent recurrent scan.  One workgroup = 32 waves on one WGP.
// Wave w owns hidden columns [16w, 16w+16).  h0/h1 live in LDS as bf16.
// Attention softmax zeroes absent rows => the next-step `keep` mask is folded
// into the per-step rescale.  GRU weights are bf16, L2-resident.
// ---------------------------------------------------------------------------
__global__ __launch_bounds__(1024) void k_recurrent(
    const int* __restrict__ y, const float* __restrict__ gi0,
    const bf16_t* __restrict__ Whh0b, const float* __restrict__ bhh0,
    const bf16_t* __restrict__ Wih1b, const bf16_t* __restrict__ Whh1b,
    const float* __restrict__ bih1, const float* __restrict__ bhh1,
    const float* __restrict__ Wihc, const float* __restrict__ Whhc,
    const float* __restrict__ bihc, const float* __restrict__ bhhc,
    const float* __restrict__ W1, const float* __restrict__ b1,
    const float* __restrict__ W2, const float* __restrict__ b2,
    const float* __restrict__ wa, const float* __restrict__ wac,
    float* __restrict__ out) {
    extern __shared__ char smem[];
    bf16_t* h0  = (bf16_t*)smem;             // [32][512] bf16 (rows 30,31 zero pad)
    bf16_t* h1  = h0 + 32 * HD;              // [32][512] bf16
    float*  hc  = (float*)(h1 + 32 * HD);    // [32][32]
    float*  hcn = hc + 32 * HC;              // [32][32]
    float*  mid = hcn + 32 * HC;             // [32][256] MLP hidden
    float*  s0  = mid + 32 * 256;            // attention scores
    float*  s1  = s0 + 32;
    float*  sc  = s1 + 32;
    float*  a0s = sc + 32;                   // attention weights
    float*  a1s = a0s + 32;
    float*  acs = a1s + 32;
    float*  pres  = acs + 32;                // present mask (this step)
    float*  oaux  = pres + 32;               // logits [30][2]
    float*  lossp = oaux + 64;               // loss accumulator

    const int tid  = threadIdx.x;
    const int lane = tid & 31;
    const int w    = tid >> 5;
    const int col0 = w * 16;
    const int ncol = col0 + (lane & 15);
    const int mh   = (lane >> 4) << 3;

    for (int i = tid; i < 32 * HD; i += 1024) { h0[i] = f2bf(0.0f); h1[i] = f2bf(0.0f); }
    for (int i = tid; i < 32 * HC; i += 1024) { hc[i] = 0.0f; hcn[i] = 0.0f; }
    if (tid == 0) lossp[0] = 0.0f;
    __syncthreads();

    for (int t = 0; t < T_STEPS; ++t) {
        // -------- P1: present flags + coordinate GRU (HC=32, pure VALU) --------
        if (tid < NBOX) pres[tid] = (y[(t * NBOX + tid) * 6 + 0] != 0) ? 1.0f : 0.0f;
        if (tid < NBOX * HC) {
            const int b = tid >> 5;
            const int j = tid & (HC - 1);
            const float nrm[4] = {1080.0f, 720.0f, 1080.0f, 720.0f};
            float cor[4];
#pragma unroll
            for (int k = 0; k < 4; ++k)
                cor[k] = (float)y[(t * NBOX + b) * 6 + 1 + k] / nrm[k];
            float ir = bihc[j], iz = bihc[HC + j], in_ = bihc[2 * HC + j];
#pragma unroll
            for (int k = 0; k < 4; ++k) {
                ir  += cor[k] * Wihc[j * 4 + k];
                iz  += cor[k] * Wihc[(HC + j) * 4 + k];
                in_ += cor[k] * Wihc[(2 * HC + j) * 4 + k];
            }
            float hr = bhhc[j], hz = bhhc[HC + j], hn = bhhc[2 * HC + j];
            for (int k = 0; k < HC; ++k) {
                const float hv = hc[b * HC + k];
                hr += hv * Whhc[j * HC + k];
                hz += hv * Whhc[(HC + j) * HC + k];
                hn += hv * Whhc[(2 * HC + j) * HC + k];
            }
            const float rr = sigm(ir + hr), zz = sigm(iz + hz);
            const float nn = tanhf(in_ + rr * hn);
            hcn[b * HC + j] = (1.0f - zz) * nn + zz * hc[b * HC + j];
        }
        __builtin_prefetch(&gi0[(size_t)(t * NBOX) * (3 * HD) + ncol], 0, 1);
        __syncthreads();

        // -------- P2: GRU layer 0 (gh = h0 @ Whh0^T via WMMA) --------
        {
            v8f accr[2], accz[2];
            accr[0] = accr[1] = accz[0] = accz[1] = v8zero();
            for (int k0 = 0; k0 < HD; k0 += 32) {
                v16bf a0 = load_a_lds(h0, 0, k0, lane);
                v16bf a1 = load_a_lds(h0, 16, k0, lane);
                v16bf br = load_b_bf(Whh0b, HD, col0, k0, lane);
                v16bf bz = load_b_bf(Whh0b, HD, HD + col0, k0, lane);
                accr[0] = wmma_bf16(a0, br, accr[0]);
                accr[1] = wmma_bf16(a1, br, accr[1]);
                accz[0] = wmma_bf16(a0, bz, accz[0]);
                accz[1] = wmma_bf16(a1, bz, accz[1]);
            }
            float rf[2][8], zf[2][8];
            const float bh_r = bhh0[ncol], bh_z = bhh0[HD + ncol];
#pragma unroll
            for (int mt = 0; mt < 2; ++mt)
#pragma unroll
                for (int i = 0; i < 8; ++i) {
                    const int m = mt * 16 + mh + i;
                    float ir = 0.0f, iz = 0.0f;
                    if (m < NBOX) {
                        const size_t gb = (size_t)(t * NBOX + m) * (3 * HD);
                        ir = gi0[gb + ncol];
                        iz = gi0[gb + HD + ncol];
                    }
                    rf[mt][i] = sigm(ir + accr[mt][i] + bh_r);
                    zf[mt][i] = sigm(iz + accz[mt][i] + bh_z);
                }
            v8f accn[2];
            accn[0] = accn[1] = v8zero();
            for (int k0 = 0; k0 < HD; k0 += 32) {
                v16bf a0 = load_a_lds(h0, 0, k0, lane);
                v16bf a1 = load_a_lds(h0, 16, k0, lane);
                v16bf bn = load_b_bf(Whh0b, HD, 2 * HD + col0, k0, lane);
                accn[0] = wmma_bf16(a0, bn, accn[0]);
                accn[1] = wmma_bf16(a1, bn, accn[1]);
            }
            const float bh_n = bhh0[2 * HD + ncol];
            float hnew[2][8];
#pragma unroll
            for (int mt = 0; mt < 2; ++mt)
#pragma unroll
                for (int i = 0; i < 8; ++i) {
                    const int m = mt * 16 + mh + i;
                    float v = 0.0f;
                    if (m < NBOX) {
                        const size_t gb = (size_t)(t * NBOX + m) * (3 * HD);
                        const float in_ = gi0[gb + 2 * HD + ncol];
                        const float nn  = tanhf(in_ + rf[mt][i] * (accn[mt][i] + bh_n));
                        const float ho  = bf2f(h0[m * HD + ncol]);
                        v = (1.0f - zf[mt][i]) * nn + zf[mt][i] * ho;
                    }
                    hnew[mt][i] = v;
                }
            __syncthreads();
#pragma unroll
            for (int mt = 0; mt < 2; ++mt)
#pragma unroll
                for (int i = 0; i < 8; ++i) {
                    const int m = mt * 16 + mh + i;
                    if (m < NBOX) h0[m * HD + ncol] = f2bf(hnew[mt][i]);
                }
            __syncthreads();
        }

        // -------- P3: GRU layer 1 (gi1 from h0_n, gh1 from h1; WMMA) --------
        {
            v8f accr[2], accz[2];
            accr[0] = accr[1] = accz[0] = accz[1] = v8zero();
            for (int k0 = 0; k0 < HD; k0 += 32) {
                v16bf a00 = load_a_lds(h0, 0, k0, lane);
                v16bf a01 = load_a_lds(h0, 16, k0, lane);
                v16bf a10 = load_a_lds(h1, 0, k0, lane);
                v16bf a11 = load_a_lds(h1, 16, k0, lane);
                v16bf bri = load_b_bf(Wih1b, HD, col0, k0, lane);
                v16bf brh = load_b_bf(Whh1b, HD, col0, k0, lane);
                v16bf bzi = load_b_bf(Wih1b, HD, HD + col0, k0, lane);
                v16bf bzh = load_b_bf(Whh1b, HD, HD + col0, k0, lane);
                accr[0] = wmma_bf16(a00, bri, accr[0]);
                accr[0] = wmma_bf16(a10, brh, accr[0]);
                accr[1] = wmma_bf16(a01, bri, accr[1]);
                accr[1] = wmma_bf16(a11, brh, accr[1]);
                accz[0] = wmma_bf16(a00, bzi, accz[0]);
                accz[0] = wmma_bf16(a10, bzh, accz[0]);
                accz[1] = wmma_bf16(a01, bzi, accz[1]);
                accz[1] = wmma_bf16(a11, bzh, accz[1]);
            }
            float rf[2][8], zf[2][8];
            const float bi_r = bih1[ncol] + bhh1[ncol];
            const float bi_z = bih1[HD + ncol] + bhh1[HD + ncol];
#pragma unroll
            for (int mt = 0; mt < 2; ++mt)
#pragma unroll
                for (int i = 0; i < 8; ++i) {
                    rf[mt][i] = sigm(accr[mt][i] + bi_r);
                    zf[mt][i] = sigm(accz[mt][i] + bi_z);
                }
            v8f ani[2], anh[2];
            ani[0] = ani[1] = anh[0] = anh[1] = v8zero();
            for (int k0 = 0; k0 < HD; k0 += 32) {
                v16bf a00 = load_a_lds(h0, 0, k0, lane);
                v16bf a01 = load_a_lds(h0, 16, k0, lane);
                v16bf a10 = load_a_lds(h1, 0, k0, lane);
                v16bf a11 = load_a_lds(h1, 16, k0, lane);
                v16bf bni = load_b_bf(Wih1b, HD, 2 * HD + col0, k0, lane);
                v16bf bnh = load_b_bf(Whh1b, HD, 2 * HD + col0, k0, lane);
                ani[0] = wmma_bf16(a00, bni, ani[0]);
                ani[1] = wmma_bf16(a01, bni, ani[1]);
                anh[0] = wmma_bf16(a10, bnh, anh[0]);
                anh[1] = wmma_bf16(a11, bnh, anh[1]);
            }
            const float bin_ = bih1[2 * HD + ncol];
            const float bhn_ = bhh1[2 * HD + ncol];
            float hnew[2][8];
#pragma unroll
            for (int mt = 0; mt < 2; ++mt)
#pragma unroll
                for (int i = 0; i < 8; ++i) {
                    const int m = mt * 16 + mh + i;
                    float v = 0.0f;
                    if (m < NBOX) {
                        const float nn = tanhf(ani[mt][i] + bin_ +
                                               rf[mt][i] * (anh[mt][i] + bhn_));
                        const float ho = bf2f(h1[m * HD + ncol]);
                        v = (1.0f - zf[mt][i]) * nn + zf[mt][i] * ho;
                    }
                    hnew[mt][i] = v;
                }
            __syncthreads();
#pragma unroll
            for (int mt = 0; mt < 2; ++mt)
#pragma unroll
                for (int i = 0; i < 8; ++i) {
                    const int m = mt * 16 + mh + i;
                    if (m < NBOX) h1[m * HD + ncol] = f2bf(hnew[mt][i]);
                }
            __syncthreads();
        }

        // -------- P4: MLP head, logits, masked outs, NLL --------
        for (int job = tid; job < NBOX * 256; job += 1024) {
            const int b = job >> 8;
            const int c = job & 255;
            float s = b1[c];
            for (int k = 0; k < HD; ++k) s += bf2f(h1[b * HD + k]) * W1[k * 256 + c];
            for (int k = 0; k < HC; ++k) s += hcn[b * HC + k] * W1[(HD + k) * 256 + c];
            mid[b * 256 + c] = s > 0.0f ? s : 0.0f;
        }
        __syncthreads();
        if (tid < NBOX * 2) {
            const int b = tid >> 1, cls = tid & 1;
            float s = b2[cls];
            for (int c = 0; c < 256; ++c) s += mid[b * 256 + c] * W2[c * 2 + cls];
            oaux[tid] = s;
            out[1 + (size_t)t * (NBOX * 2) + tid] = s * pres[b];
        }
        __syncthreads();
        if (tid == 0) {
            float L = 0.0f;
            for (int b = 0; b < NBOX; ++b)
                if (pres[b] > 0.5f) {
                    const float o0 = oaux[b * 2], o1 = oaux[b * 2 + 1];
                    const float mx = fmaxf(o0, o1);
                    const float lz = mx + logf(expf(o0 - mx) + expf(o1 - mx));
                    const int tgt = y[(t * NBOX + b) * 6 + 5];
                    L += lz - (tgt ? o1 : o0);
                }
            lossp[0] += L;
        }

        // -------- P5: attention scores s_b = tanh(h_b) . w  (wave per row) ----
        if (w < NBOX) {
            const int b = w;
            float p0 = 0.0f, p1 = 0.0f;
            for (int j = lane; j < HD; j += 32) {
                const float wj = wa[j];
                p0 += tanhf(bf2f(h0[b * HD + j])) * wj;
                p1 += tanhf(bf2f(h1[b * HD + j])) * wj;
            }
            float pc = tanhf(hcn[b * HC + lane]) * wac[lane];
#pragma unroll
            for (int off = 16; off > 0; off >>= 1) {
                p0 += __shfl_xor(p0, off, 32);
                p1 += __shfl_xor(p1, off, 32);
                pc += __shfl_xor(pc, off, 32);
            }
            if (lane == 0) { s0[b] = p0; s1[b] = p1; sc[b] = pc; }
        }
        __syncthreads();

        // -------- P6: masked softmax over present tracks --------
        if (tid == 0) {
            float* sv[3] = {s0, s1, sc};
            float* av[3] = {a0s, a1s, acs};
            for (int v = 0; v < 3; ++v) {
                float mx = -1e30f;
                for (int b = 0; b < NBOX; ++b)
                    if (pres[b] > 0.5f) mx = fmaxf(mx, sv[v][b]);
                float sum = 0.0f;
                for (int b = 0; b < NBOX; ++b) {
                    const float e = (pres[b] > 0.5f) ? expf(sv[v][b] - mx) : 0.0f;
                    av[v][b] = e;
                    sum += e;
                }
                const float d = fmaxf(sum, 1e-9f);
                for (int b = 0; b < NBOX; ++b) av[v][b] /= d;
            }
        }
        __syncthreads();

        // -------- P7: rescale states (attention * implicit keep-mask) --------
        for (int i = tid; i < NBOX * HD; i += 1024) {
            const int b = i >> 9;
            h0[i] = f2bf(bf2f(h0[i]) * a0s[b]);
            h1[i] = f2bf(bf2f(h1[i]) * a1s[b]);
        }
        for (int i = tid; i < NBOX * HC; i += 1024) {
            const int b = i >> 5;
            hc[i] = hcn[i] * acs[b];
        }
        __syncthreads();
    }
    if (tid == 0) out[0] = lossp[0];
}

// ---------------------------------------------------------------------------
extern "C" void kernel_launch(void* const* d_in, const int* in_sizes, int n_in,
                              void* d_out, int out_size, void* d_ws, size_t ws_size,
                              hipStream_t stream) {
    (void)in_sizes; (void)n_in; (void)out_size; (void)ws_size;
    const int*   y    = (const int*)  d_in[1];
    const float* flow = (const float*)d_in[2];
    const float* Wphi = (const float*)d_in[3];
    const float* bphi = (const float*)d_in[4];
    const float* Wih0 = (const float*)d_in[5];
    const float* Whh0 = (const float*)d_in[6];
    const float* bih0 = (const float*)d_in[7];
    const float* bhh0 = (const float*)d_in[8];
    const float* Wih1 = (const float*)d_in[9];
    const float* Whh1 = (const float*)d_in[10];
    const float* bih1 = (const float*)d_in[11];
    const float* bhh1 = (const float*)d_in[12];
    const float* Wihc = (const float*)d_in[13];
    const float* Whhc = (const float*)d_in[14];
    const float* bihc = (const float*)d_in[15];
    const float* bhhc = (const float*)d_in[16];
    const float* W1   = (const float*)d_in[17];
    const float* b1   = (const float*)d_in[18];
    const float* W2   = (const float*)d_in[19];
    const float* b2   = (const float*)d_in[20];
    const float* wa   = (const float*)d_in[21];
    const float* wac  = (const float*)d_in[22];

    float* xv  = (float*)d_ws;                          // [15872, 512] f32
    float* gi0 = xv + (size_t)(T_STEPS * 31) * HD;      // [15360, 1536] f32
    bf16_t* wb = (bf16_t*)(gi0 + (size_t)(T_STEPS * NBOX) * (3 * HD));
    bf16_t* WphiT = wb;                                  // [512][4096] bf16
    bf16_t* Wih0b = WphiT + (size_t)HD * XDIM;           // [1536][1024] bf16
    bf16_t* Whh0b = Wih0b + (size_t)(3 * HD) * (2 * HD); // [1536][512] bf16
    bf16_t* Wih1b = Whh0b + (size_t)(3 * HD) * HD;
    bf16_t* Whh1b = Wih1b + (size_t)(3 * HD) * HD;
    float* out = (float*)d_out;                          // [1 + 512*30*2]

    // One-time weight conversion to bf16 (K-contiguous for b128 fragment loads).
    k_prep_wphiT<<<(HD * XDIM) / 256, 256, 0, stream>>>(Wphi, WphiT);
    k_prep_bf16<<<(3 * HD * 2 * HD) / 256, 256, 0, stream>>>(Wih0, Wih0b,
                                                             3 * HD * 2 * HD);
    k_prep_bf16<<<(3 * HD * HD) / 256, 256, 0, stream>>>(Whh0, Whh0b, 3 * HD * HD);
    k_prep_bf16<<<(3 * HD * HD) / 256, 256, 0, stream>>>(Wih1, Wih1b, 3 * HD * HD);
    k_prep_bf16<<<(3 * HD * HD) / 256, 256, 0, stream>>>(Whh1, Whh1b, 3 * HD * HD);

    // xv = relu(flow @ W_phi + b): M=15872 (124x128), N=512 (8x64), K=4096.
    k_phi_gemm<<<dim3((T_STEPS * 31) / 128, HD / 64), 256, 0, stream>>>(flow, WphiT,
                                                                        bphi, xv);
    // gi0 for all timesteps: M=15360 (120x128), N=1536 (24x64), K=1024.
    k_gi0_gemm<<<dim3((T_STEPS * NBOX) / 128, (3 * HD) / 64), 256, 0, stream>>>(
        xv, Wih0b, bih0, gi0);

    const size_t shmem =
        (size_t)(2 * 32 * HD) * sizeof(bf16_t) +                       // h0, h1
        (size_t)(2 * 32 * HC + 32 * 256 + 6 * 32 + 32 + 64 + 16) * sizeof(float);
    k_recurrent<<<1, 1024, shmem, stream>>>(y, gi0, Whh0b, bhh0, Wih1b, Whh1b, bih1,
                                            bhh1, Wihc, Whhc, bihc, bhhc, W1, b1,
                                            W2, b2, wa, wac, out);
}
